// ZuoErJinYouErChu_2362232013034
// MI455X (gfx1250) — matrix operand order repaired and checked by reading, not compiled, not run on hardware
//
#include <hip/hip_runtime.h>
#include <hip/hip_bf16.h>

// ---------------------------------------------------------------------------
// B=8, S=4096, D=1024, M=32. Pipeline:
//   1) WMMA GEMM  X(32768x1024) @ WcatT(1152x1024 n-major)
//        -> gates [r|w|nm] (sigmoid/linear) + pre_x (into d_out)
//   2) chunked affine scan over S  (mem_{s+1} = (1-w) mem + w nm, rm = r*mem)
//   3) WMMA GEMM  rm(32768x32) @ Wf(32x1024) + pre_x -> tanh, in-place d_out
// CDNA5: v_wmma_f32_16x16x32_bf16 + global_load_async_to_lds_b128 staging.
// ---------------------------------------------------------------------------

typedef __attribute__((ext_vector_type(16))) __bf16        v16bf;
typedef __attribute__((ext_vector_type(8)))  float         v8f;
typedef __attribute__((ext_vector_type(4)))  unsigned int  uint4v;
typedef __attribute__((ext_vector_type(4)))  int           i4v;

union FragBF { v16bf v; uint4v u[2]; };

static constexpr int kB    = 8;
static constexpr int kS    = 4096;
static constexpr int kD    = 1024;
static constexpr int kM    = 32;
static constexpr int kBS   = kB * kS;        // 32768 rows
static constexpr int kNCAT = 3 * kM + kD;    // 1120 live columns
static constexpr int kNPAD = 1152;           // multiple of 128 (9 N-blocks)
static constexpr int kCH   = 64;             // scan chunk length
static constexpr int kNCH  = kS / kCH;       // 64 chunks per batch

// ---- CDNA5 async global->LDS staging (guarded; sync fallback) -------------
#define AS3 __attribute__((address_space(3)))
#define AS1 __attribute__((address_space(1)))

#if defined(__gfx1250__) && __has_builtin(__builtin_amdgcn_global_load_async_to_lds_b128)
#define HAVE_ASYNC 1
#else
#define HAVE_ASYNC 0
#endif

__device__ __forceinline__ AS3 i4v* lds_i4(void* p) {
  return (AS3 i4v*)(unsigned)(unsigned long long)p;   // LDS addr = low 32 bits
}
__device__ __forceinline__ AS1 i4v* glb_i4(const void* p) {
  return (AS1 i4v*)(unsigned long long)p;
}

__device__ __forceinline__ void stage16(const void* g, void* l) {
#if HAVE_ASYNC
  __builtin_amdgcn_global_load_async_to_lds_b128(glb_i4(g), lds_i4(l), 0, 0);
#else
  *(uint4v*)l = *(const uint4v*)g;
#endif
}

#if HAVE_ASYNC
#if __has_builtin(__builtin_amdgcn_s_wait_asynccnt)
#define WAIT_ASYNC(n) __builtin_amdgcn_s_wait_asynccnt(n)
#else
#define WAIT_ASYNC(n) asm volatile("s_wait_asynccnt %0" ::"n"(n) : "memory")
#endif
#else
#define WAIT_ASYNC(n)
#endif

__device__ __forceinline__ unsigned short f2bf(float x) {
  unsigned int u = __builtin_bit_cast(unsigned int, x);
  unsigned int r = u + 0x7FFFu + ((u >> 16) & 1u);   // round-to-nearest-even
  return (unsigned short)(r >> 16);
}

// --------------------------- prep kernels ----------------------------------

__global__ __launch_bounds__(256) void k_prep_x(const float* __restrict__ x,
                                                unsigned short* __restrict__ xbf) {
  size_t i = ((size_t)blockIdx.x * 256 + threadIdx.x) * 4;
  float4 v = *(const float4*)(x + i);
  uint2 p;
  p.x = (unsigned)f2bf(v.x) | ((unsigned)f2bf(v.y) << 16);
  p.y = (unsigned)f2bf(v.z) | ((unsigned)f2bf(v.w) << 16);
  *(uint2*)(xbf + i) = p;
}

// WcatT[n][k] = Wcat[k][n]; n: 0..31 Wr, 32..63 Ww, 64..95 Wm, 96..1119 Wo_x,
// 1120..1151 zero pad. Also bcat[n].
__global__ __launch_bounds__(256) void k_prep_wcat(
    const float* __restrict__ Wr, const float* __restrict__ Ww,
    const float* __restrict__ Wm, const float* __restrict__ Wo,
    const float* __restrict__ br, const float* __restrict__ bw,
    const float* __restrict__ bm, const float* __restrict__ bo,
    unsigned short* __restrict__ wcatT, float* __restrict__ bcat) {
  int n = blockIdx.x;
  for (int k = threadIdx.x; k < kD; k += 256) {
    float v;
    if      (n < 32)    v = Wr[(size_t)k * kM + n];
    else if (n < 64)    v = Ww[(size_t)k * kM + (n - 32)];
    else if (n < 96)    v = Wm[(size_t)k * kM + (n - 64)];
    else if (n < kNCAT) v = Wo[(size_t)k * kD + (n - 96)];
    else                v = 0.f;
    wcatT[(size_t)n * kD + k] = f2bf(v);
  }
  if (threadIdx.x == 0) {
    float b;
    if      (n < 32)    b = br[n];
    else if (n < 64)    b = bw[n - 32];
    else if (n < 96)    b = bm[n - 64];
    else if (n < kNCAT) b = bo[n - 96];
    else                b = 0.f;
    bcat[n] = b;
  }
}

// Wf[m][d] = sum_e Wp[m][e]*Wo_m[e][d]; row m==32 uses bp -> bfused[d].
__global__ __launch_bounds__(256) void k_prep_wf(
    const float* __restrict__ Wp, const float* __restrict__ bp,
    const float* __restrict__ Wo,
    unsigned short* __restrict__ wfT, float* __restrict__ bfused) {
  int tid = blockIdx.x * 256 + threadIdx.x;   // 33*1024 threads
  int m = tid >> 10, d = tid & 1023;
  if (m > 32) return;
  float acc = 0.f;
  for (int e = 0; e < kD; ++e) {
    float wv = (m < 32) ? Wp[(size_t)m * kD + e] : bp[e];
    acc = fmaf(wv, Wo[(size_t)(kD + e) * kD + d], acc);
  }
  if (m < 32) wfT[(size_t)d * kM + m] = f2bf(acc);
  else        bfused[d] = acc;
}

// --------------------------- main WMMA GEMM --------------------------------
// Block tile 128(M) x 128(N), 8 waves (4x2), wave tile 32x64 = 2x4 WMMA tiles.
// Double-buffered LDS fed by async global->LDS b128; one barrier per K-step.
// LDS rows padded to 80B for bank-conflict-free ds_load_b128 fragment loads.
__global__ __launch_bounds__(256) void k_gemm_main(
    const unsigned short* __restrict__ xbf,
    const unsigned short* __restrict__ wcatT,
    const float* __restrict__ bcat,
    float* __restrict__ gates,   // (BS, 96): [r|w|nm]
    float* __restrict__ prex) {  // d_out used as (BS, D) pre_x storage
  __shared__ __align__(16) unsigned char lds[2 * (128 * 80 + 128 * 80)];

  const int t = threadIdx.x;
  const int lane = t & 31, wave = t >> 5;
  const int h = (lane >> 4) & 1, lc = lane & 15;
  const int wm = wave & 3, wn = wave >> 2;
  const int m0 = blockIdx.y * 128;
  const int n0 = blockIdx.x * 128;

  // stage one K-slab (A:128x32, B:128x32 bf16) into buffer `buf`
  auto issue = [&](int kk, int buf) {
    unsigned char* la = lds + buf * 20480;
    unsigned char* lb = la + 10240;
#pragma unroll
    for (int r = 0; r < 2; ++r) {           // A: 512 x b128 -> 2/thread
      int idx = t + r * 256;
      int row = idx >> 2, seg = idx & 3;
      stage16(xbf + (size_t)(m0 + row) * kD + kk + seg * 8,
              la + row * 80 + seg * 16);
    }
#pragma unroll
    for (int r = 0; r < 2; ++r) {           // B: 512 x b128 -> 2/thread
      int idx = t + r * 256;
      int row = idx >> 2, seg = idx & 3;
      stage16(wcatT + (size_t)(n0 + row) * kD + kk + seg * 8,
              lb + row * 80 + seg * 16);
    }
  };

  v8f acc[2][4] = {};
  issue(0, 0);
  for (int ks = 0; ks < kD / 32; ++ks) {
    const int cur = ks & 1;
    WAIT_ASYNC(0);          // buffer `cur` fully written (this wave's DMAs)
    __syncthreads();        // ...and everyone's; prior reads of `nxt` done
    if (ks + 1 < kD / 32) issue((ks + 1) * 32, cur ^ 1);   // overlaps compute

    unsigned char* la = lds + cur * 20480;
    unsigned char* lb = la + 10240;
    FragBF a[2], b[4];
#pragma unroll
    for (int mi = 0; mi < 2; ++mi) {        // A frag: K = 16h..16h+15
      int ml = wm * 32 + mi * 16 + lc;
      a[mi].u[0] = *(const uint4v*)(la + ml * 80 + 32 * h);
      a[mi].u[1] = *(const uint4v*)(la + ml * 80 + 32 * h + 16);
    }
#pragma unroll
    for (int nj = 0; nj < 4; ++nj) {        // B frag: K = 16h..16h+15
      int nl = wn * 64 + nj * 16 + lc;
      b[nj].u[0] = *(const uint4v*)(lb + nl * 80 + 32 * h);
      b[nj].u[1] = *(const uint4v*)(lb + nl * 80 + 32 * h + 16);
    }
#pragma unroll
    for (int mi = 0; mi < 2; ++mi)
#pragma unroll
      for (int nj = 0; nj < 4; ++nj)
        acc[mi][nj] = __builtin_amdgcn_wmma_f32_16x16x32_bf16(
            false, a[mi].v, false, b[nj].v, (short)0, acc[mi][nj], false, false);
  }

  // Epilogue: C/D layout -> row = v + 8h, col = lane%16.
#pragma unroll
  for (int mi = 0; mi < 2; ++mi)
#pragma unroll
    for (int nj = 0; nj < 4; ++nj) {
      int col = n0 + wn * 64 + nj * 16 + lc;
#pragma unroll
      for (int v = 0; v < 8; ++v) {
        int row = m0 + wm * 32 + mi * 16 + v + 8 * h;
        float val = acc[mi][nj][v] + bcat[col];
        if (col < 64) {                       // r, w gates -> sigmoid
          gates[(size_t)row * 96 + col] = 1.f / (1.f + __expf(-val));
        } else if (col < 96) {                // new_mem
          gates[(size_t)row * 96 + col] = val;
        } else if (col < kNCAT) {             // pre_x
          prex[(size_t)row * kD + (col - 96)] = val;
        }
      }
    }
}

// --------------------------- scan (3 small kernels) ------------------------
// mem_{s+1} = a_s*mem_s + b_s with a = 1-w, b = w*nm.  rm_s = r_s * mem_s.

__global__ __launch_bounds__(32) void k_scan_composite(
    const float* __restrict__ gates, float* __restrict__ AB) {
  int cb = blockIdx.x;               // 0..511 = b*64 + chunk
  int b = cb >> 6, c = cb & 63;
  int m = threadIdx.x;
  float A = 1.f, Bc = 0.f;
  for (int i = 0; i < kCH; ++i) {
    size_t base = ((size_t)b * kS + (size_t)c * kCH + i) * 96;
    float w  = gates[base + 32 + m];
    float nm = gates[base + 64 + m];
    float a  = 1.f - w;
    A  = a * A;
    Bc = a * Bc + w * nm;
  }
  AB[((size_t)cb * 2 + 0) * kM + m] = A;
  AB[((size_t)cb * 2 + 1) * kM + m] = Bc;
}

__global__ __launch_bounds__(32) void k_scan_chunks(
    const float* __restrict__ AB, float* __restrict__ mstart) {
  int b = blockIdx.x, m = threadIdx.x;
  float mem = 0.f;
  for (int c = 0; c < kNCH; ++c) {
    int cb = b * kNCH + c;
    mstart[(size_t)cb * kM + m] = mem;
    float A  = AB[((size_t)cb * 2 + 0) * kM + m];
    float Bc = AB[((size_t)cb * 2 + 1) * kM + m];
    mem = A * mem + Bc;
  }
}

__global__ __launch_bounds__(32) void k_scan_emit(
    const float* __restrict__ gates, const float* __restrict__ mstart,
    unsigned short* __restrict__ rmbf) {
  int cb = blockIdx.x;
  int b = cb >> 6, c = cb & 63;
  int m = threadIdx.x;
  float mem = mstart[(size_t)cb * kM + m];
  for (int i = 0; i < kCH; ++i) {
    size_t s = (size_t)b * kS + (size_t)c * kCH + i;
    size_t base = s * 96;
    float r  = gates[base + m];
    float w  = gates[base + 32 + m];
    float nm = gates[base + 64 + m];
    rmbf[s * kM + m] = f2bf(r * mem);
    mem = (1.f - w) * mem + w * nm;
  }
}

// --------------------------- final WMMA GEMM (K=32) ------------------------
// out[bs,d] = tanh(pre_x + (rm @ Wf) + bfused), in-place on d_out.
// Block tile 128x128, single K=32 slab -> one WMMA per subtile.
__global__ __launch_bounds__(256) void k_gemm_final(
    const unsigned short* __restrict__ rmbf,
    const unsigned short* __restrict__ wfT,
    const float* __restrict__ bfused,
    float* __restrict__ out) {
  __shared__ __align__(16) unsigned char lds[128 * 80 + 128 * 80];
  unsigned char* la = lds;
  unsigned char* lb = lds + 10240;

  const int t = threadIdx.x;
  const int lane = t & 31, wave = t >> 5;
  const int h = (lane >> 4) & 1, lc = lane & 15;
  const int wm = wave & 3, wn = wave >> 2;
  const int m0 = blockIdx.y * 128;
  const int n0 = blockIdx.x * 128;

#pragma unroll
  for (int r = 0; r < 2; ++r) {
    int idx = t + r * 256;
    int row = idx >> 2, seg = idx & 3;
    stage16(rmbf + (size_t)(m0 + row) * kM + seg * 8, la + row * 80 + seg * 16);
  }
#pragma unroll
  for (int r = 0; r < 2; ++r) {
    int idx = t + r * 256;
    int row = idx >> 2, seg = idx & 3;
    stage16(wfT + (size_t)(n0 + row) * kM + seg * 8, lb + row * 80 + seg * 16);
  }
  WAIT_ASYNC(0);
  __syncthreads();

  v8f acc[2][4] = {};
  FragBF a[2], b[4];
#pragma unroll
  for (int mi = 0; mi < 2; ++mi) {
    int ml = wm * 32 + mi * 16 + lc;
    a[mi].u[0] = *(const uint4v*)(la + ml * 80 + 32 * h);
    a[mi].u[1] = *(const uint4v*)(la + ml * 80 + 32 * h + 16);
  }
#pragma unroll
  for (int nj = 0; nj < 4; ++nj) {
    int nl = wn * 64 + nj * 16 + lc;
    b[nj].u[0] = *(const uint4v*)(lb + nl * 80 + 32 * h);
    b[nj].u[1] = *(const uint4v*)(lb + nl * 80 + 32 * h + 16);
  }
#pragma unroll
  for (int mi = 0; mi < 2; ++mi)
#pragma unroll
    for (int nj = 0; nj < 4; ++nj)
      acc[mi][nj] = __builtin_amdgcn_wmma_f32_16x16x32_bf16(
          false, a[mi].v, false, b[nj].v, (short)0, acc[mi][nj], false, false);

#pragma unroll
  for (int mi = 0; mi < 2; ++mi)
#pragma unroll
    for (int nj = 0; nj < 4; ++nj) {
      int col = n0 + wn * 64 + nj * 16 + lc;
#pragma unroll
      for (int v = 0; v < 8; ++v) {
        int row = m0 + wm * 32 + mi * 16 + v + 8 * h;
        size_t idx = (size_t)row * kD + col;
        float val = acc[mi][nj][v] + bfused[col] + out[idx];
        out[idx] = tanhf(val);
      }
    }
}

// --------------------------- launch ----------------------------------------

extern "C" void kernel_launch(void* const* d_in, const int* in_sizes, int n_in,
                              void* d_out, int out_size, void* d_ws, size_t ws_size,
                              hipStream_t stream) {
  (void)in_sizes; (void)n_in; (void)out_size; (void)ws_size;
  const float* x  = (const float*)d_in[0];
  const float* Wr = (const float*)d_in[1];
  const float* br = (const float*)d_in[2];
  const float* Ww = (const float*)d_in[3];
  const float* bw = (const float*)d_in[4];
  const float* Wm = (const float*)d_in[5];
  const float* bm = (const float*)d_in[6];
  const float* Wp = (const float*)d_in[7];
  const float* bp = (const float*)d_in[8];
  const float* Wo = (const float*)d_in[9];
  const float* bo = (const float*)d_in[10];
  float* out = (float*)d_out;

  size_t off = 0;
  auto alloc = [&](size_t bytes) {
    void* p = (char*)d_ws + off;
    off += (bytes + 255) & ~(size_t)255;
    return p;
  };
  unsigned short* xbf    = (unsigned short*)alloc((size_t)kBS * kD * 2);   // 64 MB
  unsigned short* wcatT  = (unsigned short*)alloc((size_t)kNPAD * kD * 2); // 2.25 MB
  float*          bcat   = (float*)alloc((size_t)kNPAD * 4);
  unsigned short* wfT    = (unsigned short*)alloc((size_t)kD * kM * 2);
  float*          bfused = (float*)alloc((size_t)kD * 4);
  float*          gates  = (float*)alloc((size_t)kBS * 96 * 4);            // 12 MB
  float*          AB     = (float*)alloc((size_t)kB * kNCH * 2 * kM * 4);
  float*          mstart = (float*)alloc((size_t)kB * kNCH * kM * 4);
  unsigned short* rmbf   = (unsigned short*)alloc((size_t)kBS * kM * 2);   // 2 MB

  k_prep_x<<<(kBS * (size_t)kD) / (256 * 4), 256, 0, stream>>>(x, xbf);
  k_prep_wcat<<<kNPAD, 256, 0, stream>>>(Wr, Ww, Wm, Wo, br, bw, bm, bo, wcatT, bcat);
  k_prep_wf<<<(33 * kD + 255) / 256, 256, 0, stream>>>(Wp, bp, Wo, wfT, bfused);

  k_gemm_main<<<dim3(kNPAD / 128, kBS / 128), 256, 0, stream>>>(xbf, wcatT, bcat, gates, out);

  k_scan_composite<<<kB * kNCH, 32, 0, stream>>>(gates, AB);
  k_scan_chunks<<<kB, 32, 0, stream>>>(AB, mstart);
  k_scan_emit<<<kB * kNCH, 32, 0, stream>>>(gates, mstart, rmbf);

  k_gemm_final<<<dim3(kD / 128, kBS / 128), 256, 0, stream>>>(rmbf, wfT, bfused, out);
}
